// HMambaEncoder_27857157881954
// MI455X (gfx1250) — matrix operand order repaired and checked by reading, not compiled, not run on hardware
//
#include <hip/hip_runtime.h>
#include <hip/hip_bf16.h>

// ---------------------------------------------------------------------------
// HMamba encoder forward for MI455X (gfx1250, wave32, WMMA + TDM).
//   B=8, L=2048, D=1152
// Stage 0: Wq/Wk fp32 -> f16 once (L2-resident f16 weights)
// Stage 1: fused q/k projection + cosine -> boundary logit
//          (TDM tensor_load_to_lds stages H rows; v_wmma_f32_16x16x32_f16)
// Stage 2: boundary decision + stable compaction (block scan)
// Stage 3: EMA over chunks + run-length scatter back to (B,L,D)
// ---------------------------------------------------------------------------

typedef _Float16 v16h __attribute__((ext_vector_type(16)));
typedef _Float16 v8h  __attribute__((ext_vector_type(8)));
typedef float    v8f  __attribute__((ext_vector_type(8)));
typedef float    f4   __attribute__((ext_vector_type(4)));

#define BB      8
#define LL      2048
#define DD      1152
#define ROWPAD  1160          // LDS row stride in halves (conflict-free)

#if defined(__has_builtin)
#if __has_builtin(__builtin_amdgcn_tensor_load_to_lds)
#define USE_TDM 1
#endif
#endif

// ---------------------------------------------------------------------------
// Kernel 0: one-time weight conversion fp32 -> f16 (layout preserved).
// ---------------------------------------------------------------------------
__global__ __launch_bounds__(256)
void hm_wcvt_kernel(const float* __restrict__ Wq,
                    const float* __restrict__ Wk,
                    _Float16* __restrict__ wq16,
                    _Float16* __restrict__ wk16)
{
    const float* src = blockIdx.y ? Wk : Wq;
    _Float16*    dst = blockIdx.y ? wk16 : wq16;
    size_t i = ((size_t)blockIdx.x * 256 + threadIdx.x) * 8;
    if (i + 8 <= (size_t)DD * DD) {
        const f4* s4 = (const f4*)(src + i);
        f4 a = s4[0];
        f4 b = s4[1];
        v8h o;
        #pragma unroll
        for (int j = 0; j < 4; ++j) { o[j] = (_Float16)a[j]; o[4 + j] = (_Float16)b[j]; }
        *(v8h*)(dst + i) = o;
    }
}

// ---------------------------------------------------------------------------
// Kernel 1: per block = one batch x 16 sequence positions, 8 waves.
//   q[m, e] = sum_d H[b, l0+m,   d] * Wq[e, d]
//   k[m, e] = sum_d H[b, l0+m+1, d] * Wk[e, d]
//   bl[b, l0+m+1] = (1 - cos(q_m, k_m)) * temp + bias
// ---------------------------------------------------------------------------
__global__ __launch_bounds__(256)
void hm_cos_bl_kernel(const float* __restrict__ H,
                      const _Float16* __restrict__ wq16,
                      const _Float16* __restrict__ wk16,
                      const float* __restrict__ logT,
                      const float* __restrict__ bias,
                      float* __restrict__ bl)
{
#if USE_TDM
    __shared__ float hs32[17 * DD];       // TDM landing buffer (fp32, packed)
#endif
    __shared__ _Float16 hA[17 * ROWPAD];  // f16, padded rows (conflict-free)
    __shared__ float sQQ[16], sKK[16], sQK[16];

    const int b    = blockIdx.x >> 7;          // 128 row-tiles per batch
    const int l0   = (blockIdx.x & 127) << 4;  // first sequence position
    const int tid  = threadIdx.x;
    const int lane = tid & 31;
    const int wave = tid >> 5;
    const int rows = (LL - l0 < 17) ? (LL - l0) : 17;   // valid rows this tile

    if (tid < 16) { sQQ[tid] = 0.f; sKK[tid] = 0.f; sQK[tid] = 0.f; }

    const float* hb = H + (size_t)b * LL * DD;

#if USE_TDM
    // ---- Tensor Data Mover: 2-D tile (1152 x rows fp32) -> LDS ------------
    if (wave == 0) {
        typedef unsigned int u32x4 __attribute__((ext_vector_type(4)));
        typedef int          i32x8 __attribute__((ext_vector_type(8)));
        typedef int          i32x4 __attribute__((ext_vector_type(4)));

        unsigned long long ga  = (unsigned long long)(uintptr_t)(hb + (size_t)l0 * DD);
        unsigned int       lds = (unsigned int)(uintptr_t)&hs32[0]; // low 32b = LDS offset

        u32x4 g0;
        g0[0] = 1u;                                          // count = 1 (valid D#)
        g0[1] = lds;                                         // lds_addr
        g0[2] = (unsigned int)(ga & 0xFFFFFFFFu);            // global_addr[31:0]
        g0[3] = (unsigned int)((ga >> 32) & 0x01FFFFFFu)     // global_addr[56:32]
              | (2u << 30);                                  // type = 2 (image)
        i32x8 g1;
        g1[0] = (int)(2u << 16);                             // data_size = 4 bytes
        g1[1] = (int)((DD & 0xFFFF) << 16);                  // tensor_dim0[15:0]
        g1[2] = (int)((LL & 0xFFFF) << 16);                  // dim0 hi | tensor_dim1[15:0]
        g1[3] = (int)(DD << 16);                             // dim1 hi | tile_dim0 = 1152
        g1[4] = rows;                                        // tile_dim1 = rows
        g1[5] = DD;                                          // tensor_dim0_stride[31:0]
        g1[6] = 0;
        g1[7] = 0;
        i32x4 z4 = {0, 0, 0, 0};                             // groups 2/3 unused (2-D)
        i32x8 z8 = {0, 0, 0, 0, 0, 0, 0, 0};
        __builtin_amdgcn_tensor_load_to_lds(g0, g1, z4, z4, z8, 0);
    }
    __builtin_amdgcn_s_wait_tensorcnt(0);
    __syncthreads();                     // publish TDM data to all 8 waves
    // repack fp32 -> f16 into the padded layout
    for (int idx = tid; idx < 17 * DD; idx += 256) {
        int row = idx / DD;
        int col = idx - row * DD;
        float v = (row < rows) ? hs32[idx] : 0.f;
        hA[row * ROWPAD + col] = (_Float16)v;
    }
#else
    // ---- manual staging: global fp32 -> LDS f16 ---------------------------
    for (int idx = tid; idx < 17 * DD; idx += 256) {
        int row = idx / DD;
        int col = idx - row * DD;
        float v = (row < rows) ? hb[(size_t)(l0 + row) * DD + col] : 0.f;
        hA[row * ROWPAD + col] = (_Float16)v;
    }
#endif
    __syncthreads();

    const int m  = lane & 15;   // A: row in tile;  B: output column
    const int hi = lane >> 4;   // half-wave selector (K split)

    float qq[8], kk[8], qk[8];
    #pragma unroll
    for (int r = 0; r < 8; ++r) { qq[r] = 0.f; kk[r] = 0.f; qk[r] = 0.f; }

    // 72 n-tiles split over 8 waves
    for (int n0 = wave * 16; n0 < DD; n0 += 128) {
        v8f cq = {};
        v8f ck = {};
        const _Float16* wqrow = wq16 + (size_t)(n0 + m) * DD + hi * 16;
        const _Float16* wkrow = wk16 + (size_t)(n0 + m) * DD + hi * 16;

        for (int k0 = 0; k0 < DD; k0 += 32) {
            // A fragments (contiguous k order, same as B):
            //   lanes 0-15:  K {k0..k0+15}
            //   lanes 16-31: K {k0+16..k0+31}
            const _Float16* r0 = &hA[ m      * ROWPAD + k0 + hi * 16];
            const _Float16* r1 = &hA[(m + 1) * ROWPAD + k0 + hi * 16];
            v16h aq, ak;
            #pragma unroll
            for (int i = 0; i < 8; ++i) {
                aq[i]     = r0[i];
                aq[i + 8] = r0[8 + i];
                ak[i]     = r1[i];
                ak[i + 8] = r1[8 + i];
            }
            // B fragments: column e = n0+m across lanes; K 0-15 in lanes
            // 0-15, K 16-31 in lanes 16-31; straight f16 b128 loads.
            v16h bq = *(const v16h*)(wqrow + k0);
            v16h bk = *(const v16h*)(wkrow + k0);

            cq = __builtin_amdgcn_wmma_f32_16x16x32_f16(
                     false, aq, false, bq, (short)0, cq, false, false);
            ck = __builtin_amdgcn_wmma_f32_16x16x32_f16(
                     false, ak, false, bk, (short)0, ck, false, false);
        }
        // per-lane partials over this n-tile (lane = one feature column e)
        #pragma unroll
        for (int r = 0; r < 8; ++r) {
            qq[r] += cq[r] * cq[r];
            kk[r] += ck[r] * ck[r];
            qk[r] += cq[r] * ck[r];
        }
    }

    // reduce the 16 feature lanes of each half-wave
    #pragma unroll
    for (int r = 0; r < 8; ++r) {
        #pragma unroll
        for (int off = 1; off < 16; off <<= 1) {
            qq[r] += __shfl_xor(qq[r], off, 32);
            kk[r] += __shfl_xor(kk[r], off, 32);
            qk[r] += __shfl_xor(qk[r], off, 32);
        }
    }
    // lanes 0 (rows 0..7) and 16 (rows 8..15) hold sums; combine 8 waves.
    if (m == 0) {
        #pragma unroll
        for (int r = 0; r < 8; ++r) {
            int row = r + 8 * hi;
            atomicAdd(&sQQ[row], qq[r]);
            atomicAdd(&sKK[row], kk[r]);
            atomicAdd(&sQK[row], qk[r]);
        }
    }
    __syncthreads();

    if (tid < 16) {
        int pos = l0 + tid;                 // cos index, valid 0..L-2
        if (pos < LL - 1) {
            float t  = expf(logT[0]);
            t        = fminf(fmaxf(t, 0.1f), 10.f);
            float nq = fmaxf(sqrtf(sQQ[tid]), 1e-12f);
            float nk = fmaxf(sqrtf(sKK[tid]), 1e-12f);
            float cs = sQK[tid] / (nq * nk);
            bl[b * LL + pos + 1] = (1.f - cs) * t + bias[0];
        }
    }
}

// ---------------------------------------------------------------------------
// Kernel 2: per batch -- boundary decision, p = sigmoid(2*bl) clamped,
// stable compaction of boundary positions to the front, boundary count.
// ---------------------------------------------------------------------------
__global__ __launch_bounds__(1024)
void hm_route_sort_kernel(const float* __restrict__ bl,
                          const unsigned char* __restrict__ mask,
                          int*   __restrict__ srt,
                          float* __restrict__ pj,
                          int*   __restrict__ nbArr)
{
    __shared__ int ssum[1024];
    const int b   = blockIdx.x;
    const int tid = threadIdx.x;
    const int t0  = tid * 2;
    const int t1  = t0 + 1;

    float bl0 = (t0 == 0) ? 10.f : bl[b * LL + t0];
    float bl1 = bl[b * LL + t1];
    int   bnd0 = (bl0 > 0.f) && (mask[b * LL + t0] != 0);
    int   bnd1 = (bl1 > 0.f) && (mask[b * LL + t1] != 0);
    // softmax([-bl, bl])[1] == sigmoid(2*bl)
    float p0 = 1.f / (1.f + expf(-2.f * bl0));
    float p1 = 1.f / (1.f + expf(-2.f * bl1));
    p0 = fminf(fmaxf(p0, 1e-4f), 1.f - 1e-4f);
    p1 = fminf(fmaxf(p1, 1e-4f), 1.f - 1e-4f);

    int s = bnd0 + bnd1;
    ssum[tid] = s;
    __syncthreads();
    for (int off = 1; off < 1024; off <<= 1) {
        int v = (tid >= off) ? ssum[tid - off] : 0;
        __syncthreads();
        ssum[tid] += v;
        __syncthreads();
    }
    const int nb  = ssum[1023];
    const int pre = ssum[tid] - s;          // boundaries strictly before t0

    int d0   = bnd0 ? pre : nb + (t0 - pre);
    int pre1 = pre + bnd0;
    int d1   = bnd1 ? pre1 : nb + (t1 - pre1);

    srt[b * LL + d0] = t0;  pj[b * LL + d0] = p0;
    srt[b * LL + d1] = t1;  pj[b * LL + d1] = p1;
    if (tid == 0) nbArr[b] = nb;
}

// ---------------------------------------------------------------------------
// Kernel 3: EMA over boundary chunks + run-length expand.
//   h_0 = x_0; h_j = h_{j-1} + p_j (x_j - h_{j-1}); out[t] = h_{chunk(t)}
// chunk_idx is monotone, so chunk j covers t in [srt[j], srt[j+1}).
// ---------------------------------------------------------------------------
__global__ __launch_bounds__(128)
void hm_dechunk_kernel(const float* __restrict__ H,
                       const int*   __restrict__ srt,
                       const float* __restrict__ pj,
                       const int*   __restrict__ nbArr,
                       float* __restrict__ out)
{
    const int b  = blockIdx.y;
    const int d  = blockIdx.x * 128 + threadIdx.x;  // 9 * 128 = 1152
    const int nb = nbArr[b];

    const float* hb = H   + (size_t)b * LL * DD;
    float*       ob = out + (size_t)b * LL * DD;

    float h = 0.f;
    for (int j = 0; j < nb; ++j) {
        int t0   = srt[b * LL + j];
        int tend = (j + 1 < nb) ? srt[b * LL + j + 1] : LL;
        if (tend < LL)  // prefetch next boundary row (global_prefetch_b8)
            __builtin_prefetch(&hb[(size_t)tend * DD + d], 0, 3);
        float x = hb[(size_t)t0 * DD + d];
        float p = pj[b * LL + j];
        h = (j == 0) ? x : fmaf(p, x - h, h);
        for (int t = t0; t < tend; ++t)
            ob[(size_t)t * DD + d] = h;
    }
}

// ---------------------------------------------------------------------------
extern "C" void kernel_launch(void* const* d_in, const int* in_sizes, int n_in,
                              void* d_out, int out_size, void* d_ws, size_t ws_size,
                              hipStream_t stream)
{
    const float*         H    = (const float*)d_in[0];          // (B,L,D)
    const float*         Wq   = (const float*)d_in[1];          // (D,D)
    const float*         Wk   = (const float*)d_in[2];          // (D,D)
    const float*         logT = (const float*)d_in[3];          // ()
    const float*         bias = (const float*)d_in[4];          // ()
    const unsigned char* mask = (const unsigned char*)d_in[5];  // (B,L) bool
    float*               out  = (float*)d_out;                  // (B,L,D)

    char*  ws  = (char*)d_ws;
    float* bl  = (float*)(ws);                                   // B*L f32
    int*   srt = (int*)  (ws + (size_t)BB * LL * 4);             // B*L i32
    float* pj  = (float*)(ws + (size_t)BB * LL * 8);             // B*L f32
    int*   nbA = (int*)  (ws + (size_t)BB * LL * 12);            // B   i32
    _Float16* wq16 = (_Float16*)(ws + (size_t)BB * LL * 12 + 256);
    _Float16* wk16 = wq16 + (size_t)DD * DD;

    hm_wcvt_kernel<<<dim3((DD * DD / 8 + 255) / 256, 2), dim3(256), 0, stream>>>(
        Wq, Wk, wq16, wk16);
    hm_cos_bl_kernel<<<dim3(BB * 128), dim3(256), 0, stream>>>(
        H, wq16, wk16, logT, bias, bl);
    hm_route_sort_kernel<<<dim3(BB), dim3(1024), 0, stream>>>(
        bl, mask, srt, pj, nbA);
    hm_dechunk_kernel<<<dim3(9, BB), dim3(128), 0, stream>>>(
        H, srt, pj, nbA, out);
}
